// ComplexDotLayer_721554506371
// MI455X (gfx1250) — compile-verified, matrix operand order checked by reading, not run on hardware
//
#include <hip/hip_runtime.h>
#include <hip/hip_bf16.h>

typedef __attribute__((ext_vector_type(8)))  __bf16 v8bf;
typedef __attribute__((ext_vector_type(16))) __bf16 v16bf;
typedef __attribute__((ext_vector_type(8)))  float  v8f;

#define BN_    64
#define HWN    1024
#define CN     256
#define ON     512
#define OTILE  128
// interleaved LDS row pitches (bf16 elements)
#define XP     536   // [b]: xr[0..263], xi[264..527], pad -> 1072B rows, 12-dw bank step
#define WP     792   // [o]: wr[0..263], wi[264..527], wrn[528..791] -> 1584B rows, 12-dw bank step

static __device__ __forceinline__ __bf16 f2bf(float f) {
    unsigned u = __builtin_bit_cast(unsigned, f);
    u = u + 0x7FFFu + ((u >> 16) & 1u);             // round-to-nearest-even
    unsigned short h = (unsigned short)(u >> 16);
    return __builtin_bit_cast(__bf16, h);
}

static __device__ __forceinline__ v16bf ldfrag(const __bf16* p) {
    v8bf lo = *reinterpret_cast<const v8bf*>(p);
    v8bf hi = *reinterpret_cast<const v8bf*>(p + 16);
    return __builtin_shufflevector(lo, hi, 0,1,2,3,4,5,6,7,8,9,10,11,12,13,14,15);
}

__global__ __launch_bounds__(512)
void complex_dot_wmma(const float* __restrict__ xr, const float* __restrict__ xi,
                      const float* __restrict__ wt,
                      const float* __restrict__ gr, const float* __restrict__ br,
                      const float* __restrict__ mr, const float* __restrict__ vr,
                      const float* __restrict__ gi, const float* __restrict__ bi,
                      const float* __restrict__ mi, const float* __restrict__ vi,
                      float* __restrict__ outr, float* __restrict__ outi) {
    extern __shared__ __bf16 smem[];
    __bf16* Xl = smem;                   // 64  rows x XP
    __bf16* Wl = smem + 64 * XP;         // 128 rows x WP

    const int hw  = blockIdx.x;          // h*32 + w
    const int u   = hw >> 5;
    const int v   = hw & 31;
    const int tid = threadIdx.x;

    // ---- twiddles for the 9-tap DFT: e^{-i 2pi (u p + v q)/32} ----
    float cs[9], sn[9];
#pragma unroll
    for (int t = 0; t < 9; ++t) {
        const int p = t / 3, q = t % 3;
        const float ang = 0.19634954084936207f * (float)((u * p + v * q) & 31); // 2pi/32
        __sincosf(ang, &sn[t], &cs[t]);
    }

    // ---- stage x_real / x_imag (64 x 256) into LDS rows [xr | xi] ----
#pragma unroll 4
    for (int it = 0; it < 32; ++it) {
        const int idx = it * 512 + tid;
        const int b = idx >> 8, c = idx & 255;
        const size_t g = ((size_t)b * HWN + hw) * CN + c;
        Xl[b * XP + c]       = f2bf(xr[g]);
        Xl[b * XP + 264 + c] = f2bf(xi[g]);
    }

    // ---- generate frequency-domain weight tile, rows [wr | wi | -wr] ----
    {
        const int oL = tid & 127;
        const int c0 = tid >> 7;                      // 0..3
        const int og = blockIdx.y * OTILE + oL;
        const float* wp = wt + og;
        for (int it = 0; it < 64; ++it) {
            const int c = c0 + (it << 2);
            float wre = 0.f, wim = 0.f;
#pragma unroll
            for (int t = 0; t < 9; ++t) {
                const float w = wp[(size_t)(t * CN + c) * ON];
                wre = fmaf(w,  cs[t], wre);
                wim = fmaf(w, -sn[t], wim);
            }
            const int a = oL * WP + c;
            Wl[a]       = f2bf(wre);
            Wl[a + 264] = f2bf(wim);
            Wl[a + 528] = f2bf(-wre);
        }
    }
    __syncthreads();

    // ---- WMMA GEMM: M=64(batch) x N=128(Otile) x K=256(C) ----
    // 16 waves: wave&3 -> M-tile (16 rows), wave>>2 -> N-group of 32 (2 N-tiles)
    const int wave  = tid >> 5;
    const int lane  = tid & 31;
    const int l15   = lane & 15;
    const int lh    = lane >> 4;                       // lane-half: K offset 0/8
    const int mbase = (wave & 3) << 4;
    const int nbase = (wave >> 2) << 5;

    const __bf16* arow = Xl + (mbase + l15) * XP + (lh << 3);
    const __bf16* brow = Wl + (nbase + l15) * WP + (lh << 3);

    v8f accR[2], accI[2];
#pragma unroll
    for (int nt = 0; nt < 2; ++nt)
#pragma unroll
        for (int j = 0; j < 8; ++j) { accR[nt][j] = 0.f; accI[nt][j] = 0.f; }

    // Software pipeline: 16 steps (8 K-chunks x 2 N-tiles); while the WMMAs of
    // step s run, the fragment loads of step s+1 are in flight (double buffers:
    // B by step parity, A by K-chunk parity) -> partial dscnt waits, no drain.
    v16bf ArB[2], AiB[2], BrB[2], BiB[2], BnB[2];
    ArB[0] = ldfrag(arow);
    AiB[0] = ldfrag(arow + 264);
    BrB[0] = ldfrag(brow);
    BiB[0] = ldfrag(brow + 264);
    BnB[0] = ldfrag(brow + 528);

#pragma unroll
    for (int s = 0; s < 16; ++s) {
        const int p  = s & 1;            // B buffer parity
        const int ap = (s >> 1) & 1;     // A buffer parity
        if (s + 1 < 16) {                // prefetch next step's fragments
            const int sn  = s + 1;
            const int pn  = sn & 1;
            const int k8n = sn >> 1;
            const int ntn = sn & 1;
            const int kcn = k8n << 5;
            if (ntn == 0) {
                ArB[k8n & 1] = ldfrag(arow + kcn);
                AiB[k8n & 1] = ldfrag(arow + kcn + 264);
            }
            const __bf16* bp = brow + ntn * (16 * WP) + kcn; // imm-range offsets
            BrB[pn] = ldfrag(bp);
            BiB[pn] = ldfrag(bp + 264);
            BnB[pn] = ldfrag(bp + 528);
        }
        const int nt = s & 1;
        accR[nt] = __builtin_amdgcn_wmma_f32_16x16x32_bf16(false, ArB[ap], false, BrB[p],
                    (short)0, accR[nt], false, false);
        accI[nt] = __builtin_amdgcn_wmma_f32_16x16x32_bf16(false, ArB[ap], false, BiB[p],
                    (short)0, accI[nt], false, false);
        accR[nt] = __builtin_amdgcn_wmma_f32_16x16x32_bf16(false, AiB[ap], false, BiB[p],
                    (short)0, accR[nt], false, false);
        accI[nt] = __builtin_amdgcn_wmma_f32_16x16x32_bf16(false, AiB[ap], false, BnB[p],
                    (short)0, accI[nt], false, false);
    }

    // ---- epilogue: BN (inference) + LeakyReLU(0.2), nontemporal f32 stores ----
#pragma unroll
    for (int nt = 0; nt < 2; ++nt) {
        const int o = blockIdx.y * OTILE + nbase + (nt << 4) + l15;
        const float sr = gr[o] * rsqrtf(vr[o] + 1e-3f);
        const float si = gi[o] * rsqrtf(vi[o] + 1e-3f);
        const float br_ = br[o], mr_ = mr[o];
        const float bi_ = bi[o], mi_ = mi[o];
#pragma unroll
        for (int j = 0; j < 8; ++j) {
            const int b = mbase + j + (lh << 3);       // C/D layout: lanes>=16 hold M=j+8
            const size_t off = ((size_t)b * HWN + hw) * ON + o;
            float yr = (accR[nt][j] - mr_) * sr + br_;
            yr = (yr >= 0.f) ? yr : 0.2f * yr;
            __builtin_nontemporal_store(yr, outr + off);
            float yi2 = (accI[nt][j] - mi_) * si + bi_;
            yi2 = (yi2 >= 0.f) ? yi2 : 0.2f * yi2;
            __builtin_nontemporal_store(yi2, outi + off);
        }
    }
}

extern "C" void kernel_launch(void* const* d_in, const int* in_sizes, int n_in,
                              void* d_out, int out_size, void* d_ws, size_t ws_size,
                              hipStream_t stream) {
    const float* xr = (const float*)d_in[0];
    const float* xi = (const float*)d_in[1];
    const float* wt = (const float*)d_in[2];
    const float* gr = (const float*)d_in[3];
    const float* br = (const float*)d_in[4];
    const float* mr = (const float*)d_in[5];
    const float* vr = (const float*)d_in[6];
    const float* gi = (const float*)d_in[7];
    const float* bi = (const float*)d_in[8];
    const float* mi = (const float*)d_in[9];
    const float* vi = (const float*)d_in[10];

    float* outr = (float*)d_out;
    float* outi = outr + (size_t)BN_ * HWN * ON;

    dim3 grid(HWN, ON / OTILE);          // 1024 positions x 4 O-tiles
    dim3 block(512);                     // 16 wave32
    const size_t shmem = (size_t)(64 * XP + 128 * WP) * 2; // 271360 B

    complex_dot_wmma<<<grid, block, shmem, stream>>>(
        xr, xi, wt, gr, br, mr, vr, gi, bi, mi, vi, outr, outi);
}
